// ScaledDotProductAttention_73675868996428
// MI455X (gfx1250) — matrix operand order repaired and checked by reading, not compiled, not run on hardware
//
#include <hip/hip_runtime.h>
#include <hip/hip_bf16.h>

// CDNA5 / gfx1250 WMMA types (probe-verified signatures)
typedef __attribute__((ext_vector_type(16))) __bf16 v16bf;
typedef __attribute__((ext_vector_type(8)))  __bf16 v8bf;
typedef __attribute__((ext_vector_type(8)))  float  v8f;

#define B_    4
#define S_    2048
#define D_    1024
#define NEGV  (-1.0e9f)
#define SCALE 0.03125f   // 1/sqrt(1024)

// ---------------------------------------------------------------------------
// Pre-pass 1: convert Q and K (f32) to bf16, straight layout [B,S,D].
// ---------------------------------------------------------------------------
__global__ __launch_bounds__(256)
void cvt_qk_kernel(const float* __restrict__ Q, const float* __restrict__ K,
                   __bf16* __restrict__ Qb, __bf16* __restrict__ Kb)
{
    const float* src = blockIdx.y ? K : Q;
    __bf16*      dst = blockIdx.y ? Kb : Qb;
    size_t i = ((size_t)blockIdx.x * 256 + threadIdx.x) * 8;
    v8bf o;
#pragma unroll
    for (int j = 0; j < 8; ++j) o[j] = (__bf16)src[i + j];
    *(v8bf*)&dst[i] = o;
}

// ---------------------------------------------------------------------------
// Pre-pass 2: transpose V [B,S,D] f32 -> Vt [B,D,S] bf16 (LDS-tiled 32x32).
// ---------------------------------------------------------------------------
__global__ __launch_bounds__(256)
void transpose_v_kernel(const float* __restrict__ V, __bf16* __restrict__ Vt)
{
    __shared__ float t[32][33];
    const int k0 = blockIdx.x * 32, n0 = blockIdx.y * 32, b = blockIdx.z;
    const int tx = threadIdx.x, ty = threadIdx.y;   // 32 x 8
#pragma unroll
    for (int i = 0; i < 4; ++i)
        t[ty + 8 * i][tx] = V[((size_t)b * S_ + k0 + ty + 8 * i) * D_ + n0 + tx];
    __syncthreads();
#pragma unroll
    for (int i = 0; i < 4; ++i)
        Vt[((size_t)b * D_ + n0 + ty + 8 * i) * S_ + k0 + tx] = (__bf16)t[tx][ty + 8 * i];
}

// ---------------------------------------------------------------------------
// Helpers: fragment loads (all contiguous, no conversions)
// ---------------------------------------------------------------------------
__device__ __forceinline__ v16bf load_a_frag(const __bf16* row, int koff, int hi) {
    // A (16x32 bf16): two contiguous 8-element chunks per lane
    v8bf a0 = *(const v8bf*)&row[koff + hi * 16];
    v8bf a1 = *(const v8bf*)&row[koff + hi * 16 + 8];
    return __builtin_shufflevector(a0, a1, 0, 1, 2, 3, 4, 5, 6, 7,
                                   8, 9, 10, 11, 12, 13, 14, 15);
}

// ---------------------------------------------------------------------------
// Main kernel: one workgroup (4 wave32) per (batch, 16-query-row tile).
// LDS (64KB reused): phase 1 = Q tile [16][1024] bf16 (32KB);
// after the barrier = probs [16][2048] bf16 for phase 2.
// Raw scores / softmax staging live in the attn output region (L2-resident).
// ---------------------------------------------------------------------------
__global__ __launch_bounds__(128)
void sdpa_wmma_bf16_kernel(const __bf16* __restrict__ Qb,
                           const __bf16* __restrict__ Kb,
                           const __bf16* __restrict__ Vt,
                           float* __restrict__ Out,
                           float* __restrict__ Attn)
{
    __shared__ __bf16 smem[16 * 2048];          // 64KB
    __bf16* qb = smem;                          // [16][D_]  (phase 1 only)
    __bf16* pb = smem;                          // [16][S_]  (softmax -> phase 2)

    const int qtile = blockIdx.x;       // 0..127
    const int b     = blockIdx.y;       // 0..3
    const int qbase = qtile * 16;
    const int tid   = threadIdx.x;
    const int lane  = tid & 31;
    const int w     = tid >> 5;         // wave 0..3
    const int hi    = (lane >> 4) & 1;  // lane half
    const int ln    = lane & 15;

    const int T     = qtile + 1;            // causal key tiles
    const int TP4   = T & ~15;              // tiles covered by the 4-wide loop
    const int ncols = T * 16;
    const int kend  = (ncols + 31) & ~31;

    // ---- stage Q tile into LDS with CDNA5 async global->LDS copies ----
    {
        const __bf16* gsrc = &Qb[((size_t)b * S_ + qbase) * D_];   // contiguous 32KB
        unsigned ldsbase = (unsigned)(uintptr_t)qb;
        for (int i = tid; i < 16 * D_ * 2 / 16; i += 128) {        // 16B chunks
            unsigned loff = ldsbase + i * 16;
            const __bf16* gp = gsrc + i * 8;
            asm volatile("global_load_async_to_lds_b128 %0, %1, off"
                         :: "v"(loff), "v"(gp) : "memory");
        }
        asm volatile("s_wait_asynccnt 0x0" ::: "memory");
    }
    __syncthreads();

    // ---- Phase 1: raw scores = (Q K^T)*scale, causal mask -> Attn ----
    // Main loop: each wave takes 4 key tiles -> A fragment reused 4x.
    for (int nt0 = w * 4; nt0 < TP4; nt0 += 16) {
        const size_t krow = (size_t)b * S_ + (size_t)nt0 * 16 + ln;
        __builtin_prefetch(&Kb[(krow + 64) * D_], 0, 1);

        v8f acc0 = {}, acc1 = {}, acc2 = {}, acc3 = {};
#pragma unroll 2
        for (int dc = 0; dc < D_; dc += 32) {
            v16bf fa  = load_a_frag(&qb[ln * D_], dc, hi);
            v16bf fb0 = *(const v16bf*)&Kb[(krow)      * D_ + dc + hi * 16];
            v16bf fb1 = *(const v16bf*)&Kb[(krow + 16) * D_ + dc + hi * 16];
            v16bf fb2 = *(const v16bf*)&Kb[(krow + 32) * D_ + dc + hi * 16];
            v16bf fb3 = *(const v16bf*)&Kb[(krow + 48) * D_ + dc + hi * 16];
            acc0 = __builtin_amdgcn_wmma_f32_16x16x32_bf16(false, fa, false, fb0, (short)0, acc0, false, false);
            acc1 = __builtin_amdgcn_wmma_f32_16x16x32_bf16(false, fa, false, fb1, (short)0, acc1, false, false);
            acc2 = __builtin_amdgcn_wmma_f32_16x16x32_bf16(false, fa, false, fb2, (short)0, acc2, false, false);
            acc3 = __builtin_amdgcn_wmma_f32_16x16x32_bf16(false, fa, false, fb3, (short)0, acc3, false, false);
        }
        // C/D layout: N = ln, M = v + 8*hi.
#pragma unroll
        for (int v = 0; v < 8; ++v) {
            const int r = hi * 8 + v;
            float* arow = &Attn[((size_t)b * S_ + qbase + r) * S_];
            float s;
            s = acc0[v] * SCALE; if (nt0 * 16 + ln      > qbase + r) s = NEGV; arow[nt0 * 16 + ln]      = s;
            s = acc1[v] * SCALE; if (nt0 * 16 + 16 + ln > qbase + r) s = NEGV; arow[nt0 * 16 + 16 + ln] = s;
            s = acc2[v] * SCALE; if (nt0 * 16 + 32 + ln > qbase + r) s = NEGV; arow[nt0 * 16 + 32 + ln] = s;
            s = acc3[v] * SCALE; if (nt0 * 16 + 48 + ln > qbase + r) s = NEGV; arow[nt0 * 16 + 48 + ln] = s;
        }
    }
    // Remainder tiles (T % 16), one per wave round-robin — still wave-uniform.
    for (int nt = TP4 + w; nt < T; nt += 4) {
        const size_t krow = (size_t)b * S_ + (size_t)nt * 16 + ln;
        v8f acc = {};
#pragma unroll 2
        for (int dc = 0; dc < D_; dc += 32) {
            v16bf fa = load_a_frag(&qb[ln * D_], dc, hi);
            v16bf fb = *(const v16bf*)&Kb[krow * D_ + dc + hi * 16];
            acc = __builtin_amdgcn_wmma_f32_16x16x32_bf16(false, fa, false, fb, (short)0, acc, false, false);
        }
#pragma unroll
        for (int v = 0; v < 8; ++v) {
            const int r = hi * 8 + v;
            float s = acc[v] * SCALE;
            if (nt * 16 + ln > qbase + r) s = NEGV;
            Attn[((size_t)b * S_ + qbase + r) * S_ + nt * 16 + ln] = s;
        }
    }
    __threadfence();      // phase-1 global stores visible block-wide
    __syncthreads();      // retires qb so pb can reuse the LDS

    // ---- Softmax per row (wave w owns rows 4w..4w+3) ----
    for (int r = w * 4; r < w * 4 + 4; ++r) {
        float* arow = &Attn[((size_t)b * S_ + qbase + r) * S_];
        float mx = -INFINITY;
        for (int c = lane; c < ncols; c += 32) mx = fmaxf(mx, arow[c]);
#pragma unroll
        for (int off = 16; off >= 1; off >>= 1)
            mx = fmaxf(mx, __shfl_xor(mx, off, 32));

        float sum = 0.f;
        for (int c = lane; c < ncols; c += 32) {
            float e = __expf(arow[c] - mx);
            arow[c] = e;
            sum += e;
        }
#pragma unroll
        for (int off = 16; off >= 1; off >>= 1)
            sum += __shfl_xor(sum, off, 32);

        float inv = 1.0f / sum;
        for (int c = lane; c < ncols; c += 32) {
            float p = arow[c] * inv;
            arow[c] = p;                       // final attn output (fp32)
            pb[r * S_ + c] = (__bf16)p;        // bf16 copy for attn @ V
        }
        for (int c = ncols + lane; c < S_; c += 32) arow[c] = 0.f;
        for (int c = ncols + lane; c < kend; c += 32) pb[r * S_ + c] = (__bf16)0.0f;
    }
    __syncthreads();

    // ---- Phase 2: out = attn @ V. Each wave takes 4 D tiles (A reuse 4x). ----
    for (int nd0 = w * 4; nd0 < D_ / 16; nd0 += 16) {
        const size_t vrow = (size_t)b * D_ + (size_t)nd0 * 16 + ln;
        v8f acc0 = {}, acc1 = {}, acc2 = {}, acc3 = {};
#pragma unroll 2
        for (int k0 = 0; k0 < kend; k0 += 32) {
            v16bf fa  = load_a_frag(&pb[ln * S_], k0, hi);
            v16bf fb0 = *(const v16bf*)&Vt[(vrow)      * S_ + k0 + hi * 16];
            v16bf fb1 = *(const v16bf*)&Vt[(vrow + 16) * S_ + k0 + hi * 16];
            v16bf fb2 = *(const v16bf*)&Vt[(vrow + 32) * S_ + k0 + hi * 16];
            v16bf fb3 = *(const v16bf*)&Vt[(vrow + 48) * S_ + k0 + hi * 16];
            acc0 = __builtin_amdgcn_wmma_f32_16x16x32_bf16(false, fa, false, fb0, (short)0, acc0, false, false);
            acc1 = __builtin_amdgcn_wmma_f32_16x16x32_bf16(false, fa, false, fb1, (short)0, acc1, false, false);
            acc2 = __builtin_amdgcn_wmma_f32_16x16x32_bf16(false, fa, false, fb2, (short)0, acc2, false, false);
            acc3 = __builtin_amdgcn_wmma_f32_16x16x32_bf16(false, fa, false, fb3, (short)0, acc3, false, false);
        }
#pragma unroll
        for (int v = 0; v < 8; ++v) {
            const int m = hi * 8 + v;
            float* orow = &Out[((size_t)b * S_ + qbase + m) * D_];
            orow[nd0 * 16 + ln]      = acc0[v];
            orow[nd0 * 16 + 16 + ln] = acc1[v];
            orow[nd0 * 16 + 32 + ln] = acc2[v];
            orow[nd0 * 16 + 48 + ln] = acc3[v];
        }
    }
}

// ---------------------------------------------------------------------------
// Fallback: in-loop f32->bf16 conversion, 160KB dynamic LDS (ws too small).
// ---------------------------------------------------------------------------
__global__ __launch_bounds__(128)
void sdpa_fallback_kernel(const float* __restrict__ Q, const float* __restrict__ K,
                          const float* __restrict__ V, float* __restrict__ Out,
                          float* __restrict__ Attn)
{
    extern __shared__ char smem[];
    float*  sc = (float*)smem;
    __bf16* qb = (__bf16*)(smem + 16 * S_ * sizeof(float));

    const int qtile = blockIdx.x, b = blockIdx.y, qbase = qtile * 16;
    const int tid = threadIdx.x, lane = tid & 31, w = tid >> 5;
    const int hi = (lane >> 4) & 1, ln = lane & 15;
    const int ncols = (qtile + 1) * 16, kend = (ncols + 31) & ~31;

    for (int idx = tid; idx < 16 * D_; idx += 128) {
        int r = idx >> 10, c = idx & (D_ - 1);
        qb[idx] = (__bf16)Q[((size_t)b * S_ + qbase + r) * D_ + c];
    }
    __syncthreads();

    for (int nt = w; nt <= qtile; nt += 4) {
        const int kb = nt * 16;
        v8f acc = {};
        for (int dc = 0; dc < D_; dc += 32) {
            v16bf fa, fb;
#pragma unroll
            for (int v = 0; v < 8; ++v) {
                int ka = hi * 16 + 2 * v;
                fa[2 * v]     = qb[ln * D_ + dc + ka];
                fa[2 * v + 1] = qb[ln * D_ + dc + ka + 1];
                int kk = hi * 16 + 2 * v;
                const float* kp = &K[((size_t)b * S_ + kb + ln) * D_ + dc + kk];
                fb[2 * v]     = (__bf16)kp[0];
                fb[2 * v + 1] = (__bf16)kp[1];
            }
            acc = __builtin_amdgcn_wmma_f32_16x16x32_bf16(false, fa, false, fb, (short)0, acc, false, false);
        }
#pragma unroll
        for (int v = 0; v < 8; ++v) {
            int r = hi * 8 + v, col = kb + ln;
            float s = acc[v] * SCALE;
            if (col > qbase + r) s = NEGV;
            sc[r * S_ + col] = s;
        }
    }
    __syncthreads();

    for (int r = w * 4; r < w * 4 + 4; ++r) {
        float mx = -INFINITY;
        for (int c = lane; c < ncols; c += 32) mx = fmaxf(mx, sc[r * S_ + c]);
#pragma unroll
        for (int off = 16; off >= 1; off >>= 1) mx = fmaxf(mx, __shfl_xor(mx, off, 32));
        float sum = 0.f;
        for (int c = lane; c < ncols; c += 32) {
            float e = __expf(sc[r * S_ + c] - mx); sc[r * S_ + c] = e; sum += e;
        }
#pragma unroll
        for (int off = 16; off >= 1; off >>= 1) sum += __shfl_xor(sum, off, 32);
        float inv = 1.0f / sum;
        float* arow = &Attn[((size_t)b * S_ + qbase + r) * S_];
        for (int c = lane; c < ncols; c += 32) {
            float p = sc[r * S_ + c] * inv; sc[r * S_ + c] = p; arow[c] = p;
        }
        for (int c = ncols + lane; c < S_; c += 32) arow[c] = 0.f;
        for (int c = ncols + lane; c < kend; c += 32) sc[r * S_ + c] = 0.f;
    }
    __syncthreads();

    for (int nd = w; nd < D_ / 16; nd += 4) {
        const int n0 = nd * 16;
        v8f acc = {};
        for (int k0 = 0; k0 < kend; k0 += 32) {
            v16bf fa, fb;
#pragma unroll
            for (int v = 0; v < 8; ++v) {
                int ka = hi * 16 + 2 * v;
                fa[2 * v]     = (__bf16)sc[ln * S_ + k0 + ka];
                fa[2 * v + 1] = (__bf16)sc[ln * S_ + k0 + ka + 1];
                int kk = k0 + hi * 16 + 2 * v;
                fb[2 * v]     = (__bf16)V[((size_t)b * S_ + kk)     * D_ + n0 + ln];
                fb[2 * v + 1] = (__bf16)V[((size_t)b * S_ + kk + 1) * D_ + n0 + ln];
            }
            acc = __builtin_amdgcn_wmma_f32_16x16x32_bf16(false, fa, false, fb, (short)0, acc, false, false);
        }
#pragma unroll
        for (int v = 0; v < 8; ++v) {
            int m = hi * 8 + v;
            Out[((size_t)b * S_ + qbase + m) * D_ + n0 + ln] = acc[v];
        }
    }
}

extern "C" void kernel_launch(void* const* d_in, const int* in_sizes, int n_in,
                              void* d_out, int out_size, void* d_ws, size_t ws_size,
                              hipStream_t stream) {
    (void)in_sizes; (void)n_in; (void)out_size;
    const float* Q = (const float*)d_in[0];
    const float* K = (const float*)d_in[1];
    const float* V = (const float*)d_in[2];
    // d_in[3] is the tril mask; causality is applied analytically (identical result).

    float* Out  = (float*)d_out;                   // [B,S,D]
    float* Attn = Out + (size_t)B_ * S_ * D_;      // [B,S,S]

    const size_t nElem  = (size_t)B_ * S_ * D_;            // 8M per tensor
    const size_t wsNeed = 3 * nElem * sizeof(__bf16);      // 48MB

    if (ws_size >= wsNeed) {
        __bf16* Qb = (__bf16*)d_ws;
        __bf16* Kb = Qb + nElem;
        __bf16* Vt = Kb + nElem;
        cvt_qk_kernel<<<dim3((unsigned)(nElem / (256 * 8)), 2), 256, 0, stream>>>(Q, K, Qb, Kb);
        transpose_v_kernel<<<dim3(S_ / 32, D_ / 32, B_), dim3(32, 8), 0, stream>>>(V, Vt);
        sdpa_wmma_bf16_kernel<<<dim3(S_ / 16, B_), 128, 0, stream>>>(Qb, Kb, Vt, Out, Attn);
    } else {
        const size_t shmem = 16 * S_ * sizeof(float) + 16 * D_ * sizeof(__bf16); // 160KB
        sdpa_fallback_kernel<<<dim3(S_ / 16, B_), 128, shmem, stream>>>(Q, K, V, Out, Attn);
    }
}
